// selfAttention_48593259987096
// MI455X (gfx1250) — compile-verified, matrix operand order checked by reading, not run on hardware
//
#include <hip/hip_runtime.h>

// Problem constants (B, C, H, W) = (4, 64, 192, 192), pool (4, 16)
#define BB  4
#define CC  64
#define HH  192
#define WW  192
#define NN  (HH * WW)      // 36864 queries
#define PHH 48
#define PWW 12
#define MM  576            // pooled keys/values

typedef __attribute__((ext_vector_type(16))) _Float16 v16h;
typedef __attribute__((ext_vector_type(8)))  _Float16 v8h;
typedef __attribute__((ext_vector_type(8)))  float    v8f;
typedef __attribute__((ext_vector_type(4)))  int      v4i;

// --------------------------- async copy to LDS ------------------------------
#ifdef __has_builtin
#  if __has_builtin(__builtin_amdgcn_global_load_async_to_lds_b128)
#    define USE_ASYNC_LDS 1
#  endif
#endif
#ifndef USE_ASYNC_LDS
#  define USE_ASYNC_LDS 0
#endif

// 16B global -> LDS copy. With async support this emits
// global_load_async_to_lds_b128 (ASYNCcnt-tracked, bypasses VGPRs).
// Builtin signature (from hipcc diagnostic): (v4i AS1*, v4i AS3*, Imm, Imm).
__device__ __forceinline__ void cp16_to_lds(_Float16* lds_dst, const _Float16* gsrc) {
#if USE_ASYNC_LDS
  __builtin_amdgcn_global_load_async_to_lds_b128(
      (__attribute__((address_space(1))) v4i*)(void*)gsrc,
      (__attribute__((address_space(3))) v4i*)(void*)lds_dst,
      /*imm offset=*/0, /*cpol=*/0);
#else
  *(uint4*)lds_dst = *(const uint4*)gsrc;
#endif
}

__device__ __forceinline__ void async_copies_done() {
#if USE_ASYNC_LDS
#  if __has_builtin(__builtin_amdgcn_s_wait_asynccnt)
  __builtin_amdgcn_s_wait_asynccnt(0);
#  else
  asm volatile("s_wait_asynccnt 0x0" ::: "memory");
#  endif
#endif
}

// --------------------------- WMMA fragments ---------------------------------
// 16x16x32 f16 A/B fragment (CDNA5 ISA 7.12.2): lane l holds row/col (l&15);
// halves 0..7 = K k0..k0+7, halves 8..15 = K k0+16..k0+23, k0 = (l>>4)*8.
// p points at the 32 contiguous halves of this row's K-chunk.
__device__ __forceinline__ v16h frag16(const _Float16* p, int lane) {
  const int k0 = (lane >> 4) << 3;
  v8h lo = *(const v8h*)(p + k0);
  v8h hi = *(const v8h*)(p + k0 + 16);
  return __builtin_shufflevector(lo, hi, 0, 1, 2, 3, 4, 5, 6, 7,
                                 8, 9, 10, 11, 12, 13, 14, 15);
}

__device__ __forceinline__ v8f wmma_f16(v16h a, v16h b, v8f c) {
  // (neg_a, A, neg_b, B, c_mod, C, reuse_a, reuse_b)
  return __builtin_amdgcn_wmma_f32_16x16x32_f16(false, a, false, b, (short)0, c,
                                                false, false);
}

// ---------------------------------------------------------------------------
// Kernel 1: avg-pool K and V (4x16 windows), emit Kp^T as f16 [b][m][c]
// (B-matrix friendly: the 64 C-halves of one key column are contiguous),
// Vp as f32 [b][c][m]. Also zero the softmax denominators.
// ---------------------------------------------------------------------------
__global__ void pool_kernel(const float* __restrict__ Kin,
                            const float* __restrict__ Vin,
                            _Float16* __restrict__ Kp_t,
                            float* __restrict__ Vp,
                            float* __restrict__ denom) {
  int tid = blockIdx.x * blockDim.x + threadIdx.x;
  if (tid < BB * MM) denom[tid] = 0.0f;
  if (tid >= BB * CC * MM) return;

  int m = tid % MM;
  int c = (tid / MM) % CC;
  int b = tid / (MM * CC);
  int ph = m / PWW, pw = m % PWW;

  size_t base = ((size_t)b * CC + c) * (size_t)(HH * WW) +
                (size_t)(ph * 4) * WW + (size_t)pw * 16;
  const float* kb = Kin + base;
  const float* vb = Vin + base;

  float sk = 0.0f, sv = 0.0f;
#pragma unroll
  for (int r = 0; r < 4; r++) {
    const float4* kr = (const float4*)(kb + (size_t)r * WW);
    const float4* vr = (const float4*)(vb + (size_t)r * WW);
#pragma unroll
    for (int q = 0; q < 4; q++) {
      float4 a = kr[q];
      sk += a.x + a.y + a.z + a.w;
      float4 v = vr[q];
      sv += v.x + v.y + v.z + v.w;
    }
  }
  sk *= (1.0f / 64.0f);
  sv *= (1.0f / 64.0f);

  Kp_t[((size_t)b * MM + m) * CC + c] = (_Float16)sk;
  Vp[((size_t)b * CC + c) * MM + m] = sv;
}

// ---------------------------------------------------------------------------
// Kernel 2: denom[b][m] = sum_n exp(s[n][m]),  s = Qp(NxC) x Kp(CxM).
// Block = 128 threads (4 waves); each wave owns 128 query rows (8 tiles of 16).
// Kp^T for the batch is async-staged once into LDS (73.7 KB); column sums are
// kept in 36 per-lane registers and flushed with one atomicAdd per column.
// Inner loop is software-pipelined: WMMAs of tile mt+1 overlap exp of tile mt.
// ---------------------------------------------------------------------------
__global__ void denom_kernel(const float* __restrict__ Q,
                             const _Float16* __restrict__ Kp_t,
                             float* __restrict__ denom) {
  extern __shared__ _Float16 smem[];
  _Float16* kp_s = smem;                               // MM*CC halves
  const int b = blockIdx.y;
  const int wave = threadIdx.x >> 5;
  const int lane = threadIdx.x & 31;
  _Float16* q_s = smem + MM * CC + wave * (16 * 64);   // per-wave 16x64

  // Stage Kp^T[b] into LDS (async when available).
  {
    const _Float16* src = Kp_t + (size_t)b * MM * CC;
    for (int i = threadIdx.x; i < (MM * CC) / 8; i += blockDim.x)
      cp16_to_lds(kp_s + i * 8, src + i * 8);
  }
  async_copies_done();
  __syncthreads();

  const float* Qb = Q + (size_t)b * CC * NN;
  const int nwave0 = (blockIdx.x * 4 + wave) * 128;
  const int col = lane & 15;

  float acc[36];
#pragma unroll
  for (int i = 0; i < 36; i++) acc[i] = 0.0f;

  for (int nb = 0; nb < 8; nb++) {
    const int n0 = nwave0 + nb * 16;
    // Stage Q block [16 n][64 c] as f16 (Q is stored [c][n]).
    for (int i = lane; i < 1024; i += 32) {
      int c = i >> 4, nn = i & 15;
      q_s[nn * 64 + c] = (_Float16)Qb[(size_t)c * NN + (n0 + nn)];
    }
    __syncthreads();  // uniform across all 4 waves

    const _Float16* qrow = q_s + col * 64;
    v16h a0 = frag16(qrow, lane);
    v16h a1 = frag16(qrow + 32, lane);

    auto score_tile = [&](int mt) -> v8f {
      const _Float16* kcol = kp_s + (size_t)(mt * 16 + col) * 64;
      v8f d = {0.f, 0.f, 0.f, 0.f, 0.f, 0.f, 0.f, 0.f};
      d = wmma_f16(a0, frag16(kcol, lane), d);
      d = wmma_f16(a1, frag16(kcol + 32, lane), d);
      return d;
    };

    // Software pipeline: WMMAs for mt+1 issue before the exp/sum of mt,
    // hiding the WMMA->VALU hazard and the v_exp dependency chain.
    v8f d = score_tile(0);
#pragma unroll
    for (int mt = 0; mt < 36; mt++) {
      v8f dn;
      if (mt < 35) dn = score_tile(mt + 1);
      float s = 0.0f;
#pragma unroll
      for (int g = 0; g < 8; g++) s += __expf(d[g]);
      acc[mt] += s;  // lane holds column m = mt*16 + (lane&15), 8 rows summed
      d = dn;
    }
    __syncthreads();
  }

#pragma unroll
  for (int mt = 0; mt < 36; mt++) {
    float v = acc[mt];
    v += __shfl_xor(v, 16, 32);  // lanes l and l+16 hold the same column
    if (lane < 16) atomicAdd(&denom[b * MM + mt * 16 + lane], v);
  }
}

// ---------------------------------------------------------------------------
// Kernel 3: Vs[b][c][m] = (f16)(Vp[b][c][m] / denom[b][m])
// ---------------------------------------------------------------------------
__global__ void vscale_kernel(const float* __restrict__ Vp,
                              const float* __restrict__ denom,
                              _Float16* __restrict__ Vs) {
  int tid = blockIdx.x * blockDim.x + threadIdx.x;
  if (tid >= BB * CC * MM) return;
  int m = tid % MM;
  int b = tid / (CC * MM);
  Vs[tid] = (_Float16)(Vp[tid] / denom[b * MM + m]);
}

// ---------------------------------------------------------------------------
// Kernel 4: out[b][c][n] = sum_m Vs[c][m] * exp(s[n][m]).
// Block = 128 threads (4 waves); each wave owns 16 query rows.
// Per 32-wide m-chunk: GEMM1 (Q x Kp) -> exp -> wave-private LDS transpose
// ([n][m] tile becomes GEMM2's B fragment) -> GEMM2 vs 4 c-tiles of Vs.
// Kp^T and Vs for the batch are async-staged into LDS (2 x 73.7 KB).
// The exp tile is double-buffered so GEMM1 of chunk mc+1 overlaps GEMM2 of mc.
// ---------------------------------------------------------------------------
__global__ void attn_out_kernel(const float* __restrict__ Q,
                                const _Float16* __restrict__ Kp_t,
                                const _Float16* __restrict__ Vs,
                                float* __restrict__ out) {
  extern __shared__ _Float16 smem[];
  _Float16* kp_s = smem;                 // MM*CC
  _Float16* vs_s = smem + MM * CC;       // MM*CC
  const int b = blockIdx.y;
  const int wave = threadIdx.x >> 5;
  const int lane = threadIdx.x & 31;
  _Float16* q_s = smem + 2 * MM * CC + wave * 1024;              // 16 x 64
  _Float16* e_s = smem + 2 * MM * CC + 4 * 1024 + wave * 1024;   // 2 x (16n x 32m)

  {
    const _Float16* s1 = Kp_t + (size_t)b * MM * CC;
    const _Float16* s2 = Vs + (size_t)b * CC * MM;
    for (int i = threadIdx.x; i < (MM * CC) / 8; i += blockDim.x) {
      cp16_to_lds(kp_s + i * 8, s1 + i * 8);
      cp16_to_lds(vs_s + i * 8, s2 + i * 8);
    }
  }

  const float* Qb = Q + (size_t)b * CC * NN;
  const int n0 = (blockIdx.x * 4 + wave) * 16;
  for (int i = lane; i < 1024; i += 32) {
    int c = i >> 4, nn = i & 15;
    q_s[nn * 64 + c] = (_Float16)Qb[(size_t)c * NN + (n0 + nn)];
  }
  async_copies_done();
  __syncthreads();

  const int col = lane & 15;          // fragment row/col owned by this lane
  const int gbase = (lane >> 4) << 3; // D-tile row base for this lane half

  const _Float16* qrow = q_s + col * 64;
  v16h a0 = frag16(qrow, lane);
  v16h a1 = frag16(qrow + 32, lane);

  v8f acc[4];
#pragma unroll
  for (int ct = 0; ct < 4; ct++)
    acc[ct] = (v8f){0.f, 0.f, 0.f, 0.f, 0.f, 0.f, 0.f, 0.f};

  // GEMM1 for one 32-wide key chunk: scores -> exp -> e_s[buf] as [n][m]
  // (wave-private; same-wave LDS ordering makes barriers unnecessary).
  auto gemm1_chunk = [&](int mc, int buf) {
    _Float16* eb = e_s + buf * 512;
#pragma unroll
    for (int t = 0; t < 2; t++) {
      const _Float16* kcol = kp_s + (size_t)((mc * 2 + t) * 16 + col) * 64;
      v8f d = {0.f, 0.f, 0.f, 0.f, 0.f, 0.f, 0.f, 0.f};
      d = wmma_f16(a0, frag16(kcol, lane), d);
      d = wmma_f16(a1, frag16(kcol + 32, lane), d);
      const int mloc = t * 16 + col;  // D: lane holds column m, rows n=gbase+g
#pragma unroll
      for (int g = 0; g < 8; g++)
        eb[(gbase + g) * 32 + mloc] = (_Float16)__expf(d[g]);
    }
  };

  gemm1_chunk(0, 0);
  for (int mc = 0; mc < 18; mc++) {   // 18 chunks of 32 keys
    // Load this chunk's GEMM2 fragments first ...
    const _Float16* eb = e_s + (mc & 1) * 512;
    v16h bb = frag16(eb + col * 32, lane);
    v16h va[4];
#pragma unroll
    for (int ct = 0; ct < 4; ct++)
      va[ct] = frag16(vs_s + (size_t)(ct * 16 + col) * MM + mc * 32, lane);
    // ... then overlap GEMM1 of the next chunk (other buffer) ...
    if (mc < 17) gemm1_chunk(mc + 1, (mc + 1) & 1);
    // ... and accumulate GEMM2: B = exp tile (K=32 keys x 16 queries).
#pragma unroll
    for (int ct = 0; ct < 4; ct++)
      acc[ct] = wmma_f16(va[ct], bb, acc[ct]);
  }

  float* ob = out + (size_t)b * CC * NN;
  const int ncol = n0 + col;
#pragma unroll
  for (int ct = 0; ct < 4; ct++)
#pragma unroll
    for (int g = 0; g < 8; g++)
      ob[(size_t)(ct * 16 + gbase + g) * NN + ncol] = acc[ct][g];
}

// ---------------------------------------------------------------------------
extern "C" void kernel_launch(void* const* d_in, const int* in_sizes, int n_in,
                              void* d_out, int out_size, void* d_ws, size_t ws_size,
                              hipStream_t stream) {
  const float* Kin = (const float*)d_in[0];
  const float* Qin = (const float*)d_in[1];
  const float* Vin = (const float*)d_in[2];
  float* out = (float*)d_out;

  char* ws = (char*)d_ws;
  _Float16* Kp_t = (_Float16*)(ws + 0);        // 294912 B
  float* Vp      = (float*)(ws + 294912);      // 589824 B
  _Float16* Vs   = (_Float16*)(ws + 884736);   // 294912 B
  float* denom   = (float*)(ws + 1179648);     // 9216 B   (total ~1.19 MB)

  // 1) pool + zero denominators
  pool_kernel<<<(BB * CC * MM + 255) / 256, 256, 0, stream>>>(Kin, Vin, Kp_t, Vp, denom);

  // 2) softmax denominators (per-column sum of exp over all 36864 queries)
  {
    dim3 grid(NN / 512, BB);  // wave=128 rows, 4 waves/block
    size_t sh = (size_t)(MM * CC + 4 * 1024) * sizeof(_Float16);  // 81920 B
    denom_kernel<<<grid, 128, sh, stream>>>(Qin, Kp_t, denom);
  }

  // 3) scale V by denominators, convert to f16
  vscale_kernel<<<(BB * CC * MM + 255) / 256, 256, 0, stream>>>(Vp, denom, Vs);

  // 4) fused scores -> exp -> output GEMM
  {
    dim3 grid(NN / 64, BB);   // wave=16 rows, 4 waves/block
    size_t sh = (size_t)(2 * MM * CC + 4 * 1024 + 4 * 1024) * sizeof(_Float16);  // 163840 B
    attn_out_kernel<<<grid, 128, sh, stream>>>(Qin, Kp_t, Vs, out);
  }
}
